// TextBrain_52939766890771
// MI455X (gfx1250) — matrix operand order repaired and checked by reading, not compiled, not run on hardware
//
#include <hip/hip_runtime.h>

// Problem constants (from reference): B=8, S=32, D=256, N=256, K=51, VOCAB=50257
constexpr int TB = 8;
constexpr int TS = 32;
constexpr int TD = 256;
constexpr int TN = 256;
constexpr int TK = 51;
constexpr int TV = 50257;

typedef __attribute__((ext_vector_type(16))) _Float16 v16h;
typedef __attribute__((ext_vector_type(8)))  float    v8f;

union V16H { v16h v; _Float16 h[16]; };
union V8F  { v8f v;  float    f[8];  };
union H4   { _Float16 h[4]; unsigned long long u; };

__device__ __forceinline__ float sigmoidf_(float x) { return 1.0f / (1.0f + __expf(-x)); }
__device__ __forceinline__ float gelu_(float x)     { return 0.5f * x * (1.0f + erff(x * 0.70710678118654752f)); }

// ---------------------------------------------------------------------------
// Wave-cooperative GEMM: Out[16][256] = A[16][256](f16, LDS) * W[256][256]^T
// W is row-major f32 in global ([O=256][D=256]); converted to f16 in LDS per
// 32-wide K chunk with vectorized b128 loads / b64 LDS stores + prefetch of
// the next chunk. Block = 256 threads = 8 waves; wave w computes n-tiles
// 2w, 2w+1 (columns 32w..32w+31). Fragment layouts per CDNA5 ISA 7.12.2.
// ---------------------------------------------------------------------------
__device__ __forceinline__ void gemm_xWT(const _Float16* __restrict__ Ash, // [16*TD]
                                         const float* __restrict__ Wg,    // [256*TD] f32
                                         _Float16* __restrict__ Wsh,      // [256*32]
                                         V8F c[2], int tid)
{
    const int lane = tid & 31;
    const int wv   = tid >> 5;
    const int m    = lane & 15;
    const int hl   = lane >> 4;

    for (int kb = 0; kb < TD; kb += 32) {
        __syncthreads();
        // Stage W chunk: Wsh[o][kk] = (f16)Wg[o*TD + kb + kk]
        // float4 loads (16B, coalesced) -> packed 4xf16 b64 LDS stores.
        for (int i = tid * 4; i < 256 * 32; i += 256 * 4) {
            int o  = i >> 5;
            int kk = i & 31;
            const float* src = Wg + (size_t)o * TD + kb + kk;
            if (kk == 0 && kb + 32 < TD) __builtin_prefetch(src + 32, 0, 3);
            float4 w4 = *(const float4*)src;
            H4 p;
            p.h[0] = (_Float16)w4.x; p.h[1] = (_Float16)w4.y;
            p.h[2] = (_Float16)w4.z; p.h[3] = (_Float16)w4.w;
            *(unsigned long long*)(Wsh + i) = p.u;
        }
        __syncthreads();

        // A fragment (16-bit A 16x32): VGPR v<4 -> K = 2v (+8 for hi half-wave),
        // v>=4 -> K = 16 + 2(v-4) (+8 for hi half-wave); 2 f16 per VGPR.
        V16H a;
#pragma unroll
        for (int v = 0; v < 8; ++v) {
            int k0 = ((v < 4) ? (2 * v) : (16 + 2 * (v - 4))) + (hl ? 8 : 0);
            a.h[2 * v]     = Ash[m * TD + kb + k0];
            a.h[2 * v + 1] = Ash[m * TD + kb + k0 + 1];
        }

#pragma unroll
        for (int s = 0; s < 2; ++s) {
            int nt = wv * 2 + s;
            // B fragment (16-bit B 32x16): lane n = lane&15; same K order as A fragment.
            V16H b;
#pragma unroll
            for (int v = 0; v < 8; ++v) {
                int k = ((v < 4) ? (2 * v) : (16 + 2 * (v - 4))) + (hl ? 8 : 0);
                int o = nt * 16 + (lane & 15);
                b.h[2 * v]     = Wsh[o * 32 + k];
                b.h[2 * v + 1] = Wsh[o * 32 + k + 1];
            }
            c[s].v = __builtin_amdgcn_wmma_f32_16x16x32_f16(
                false, a.v, false, b.v, (short)0, c[s].v, false, false);
        }
    }
    __syncthreads();
}

// Stage a [16][TD] f16 A tile from an f32 [TB][TD] source (rows >= TB zeroed).
__device__ __forceinline__ void stage_A_f32(const float* __restrict__ src,
                                            _Float16* __restrict__ Ash, int tid)
{
    for (int i = tid * 4; i < 16 * TD; i += 256 * 4) {
        int r = i >> 8;   // TD == 256
        H4 p;
        if (r < TB) {
            float4 x4 = *(const float4*)(src + i);
            p.h[0] = (_Float16)x4.x; p.h[1] = (_Float16)x4.y;
            p.h[2] = (_Float16)x4.z; p.h[3] = (_Float16)x4.w;
        } else {
            p.u = 0ULL;
        }
        *(unsigned long long*)(Ash + i) = p.u;
    }
}

// ---------------------------------------------------------------------------
// 1) Embedding gather: x[s][b][d] = embed[ids[b][s]][d] + pos[s][d]
// ---------------------------------------------------------------------------
__global__ __launch_bounds__(256) void embed_kernel(const int* __restrict__ ids,
                                                    const float* __restrict__ emb,
                                                    const float* __restrict__ pos,
                                                    float* __restrict__ x)
{
    int bs = blockIdx.x;          // b*TS + s
    int b = bs / TS, s = bs % TS;
    int d = threadIdx.x;
    int tok = ids[b * TS + s];
    x[((size_t)s * TB + b) * TD + d] = emb[(size_t)tok * TD + d] + pos[s * TD + d];
}

// ---------------------------------------------------------------------------
// 2) Router + top-51 + softmax + active set + edge compaction. One block per t.
// ---------------------------------------------------------------------------
__global__ __launch_bounds__(256) void router_kernel(const float* __restrict__ x,
                                                     const float* __restrict__ rW,
                                                     const float* __restrict__ rb,
                                                     const int* __restrict__ esrc,
                                                     const int* __restrict__ edst,
                                                     int E,
                                                     int* __restrict__ active,
                                                     float* __restrict__ wout,
                                                     int* __restrict__ posArr,
                                                     int* __restrict__ aeIdx,
                                                     int* __restrict__ aeCnt)
{
    int t = blockIdx.x, tid = threadIdx.x;
    __shared__ float xsh[TB * TD];
    __shared__ float ssh[TB * TN];
    __shared__ float rv[256];
    __shared__ int   ri[256];
    __shared__ float topv[TK];
    __shared__ int   topi[TK];
    __shared__ int   spos[TN];
    __shared__ int   cnt;

    for (int i = tid; i < TB * TD; i += 256) xsh[i] = x[(size_t)t * TB * TD + i];
    spos[tid] = -1;
    __syncthreads();

    // scores[b][n]: thread tid handles n = tid for all 8 batch rows.
    {
        const float* wrow = rW + (size_t)tid * TD;
        for (int b = 0; b < TB; ++b) {
            float acc = rb[tid];
            for (int d = 0; d < TD; ++d) acc += xsh[b * TD + d] * wrow[d];
            ssh[b * TN + tid] = acc;
        }
    }
    __syncthreads();

    for (int b = 0; b < TB; ++b) {
        for (int k = 0; k < TK; ++k) {
            rv[tid] = ssh[b * TN + tid];
            ri[tid] = tid;
            __syncthreads();
            for (int off = 128; off > 0; off >>= 1) {
                if (tid < off) {
                    float v2 = rv[tid + off]; int i2 = ri[tid + off];
                    if (v2 > rv[tid] || (v2 == rv[tid] && i2 < ri[tid])) { rv[tid] = v2; ri[tid] = i2; }
                }
                __syncthreads();
            }
            if (tid == 0) {
                topv[k] = rv[0];
                topi[k] = ri[0];
                ssh[b * TN + ri[0]] = -1e30f;   // remove from next rounds
            }
            __syncthreads();
        }
        // softmax over this row's top-K values (topv[0] is the max).
        if (tid == 0) {
            float mx = topv[0], sum = 0.f;
            for (int k = 0; k < TK; ++k) { float e = __expf(topv[k] - mx); topv[k] = e; sum += e; }
            float inv = 1.0f / sum;
            for (int k = 0; k < TK; ++k) wout[((size_t)t * TB + b) * TK + k] = topv[k] * inv;
        }
        __syncthreads();
        if (b == 0 && tid < TK) {
            active[t * TK + tid] = topi[tid];
            spos[topi[tid]] = tid;
        }
        __syncthreads();
    }

    posArr[t * TN + tid] = spos[tid];
    if (tid == 0) cnt = 0;
    __syncthreads();
    for (int e = tid; e < E; e += 256) {
        if (spos[esrc[e]] >= 0 && spos[edst[e]] >= 0) {
            int i = atomicAdd(&cnt, 1);
            aeIdx[(size_t)t * E + i] = e;
        }
    }
    __syncthreads();
    if (tid == 0) aeCnt[t] = cnt;
}

// ---------------------------------------------------------------------------
// 3) Hop 0: states[t][k] = gelu(xt @ fc_W[a]^T + fc_b[a]) * sigmoid(xt.gate) * w
//    One block per (k, t); WMMA GEMM [16x256] x [256x256]^T.
// ---------------------------------------------------------------------------
__global__ __launch_bounds__(256) void hop0_kernel(const float* __restrict__ x,
                                                   const float* __restrict__ fcW,
                                                   const float* __restrict__ fcb,
                                                   const float* __restrict__ gW,
                                                   const float* __restrict__ gb,
                                                   const int* __restrict__ active,
                                                   const float* __restrict__ wbuf,
                                                   float* __restrict__ states,
                                                   _Float16* __restrict__ states_h)
{
    int t = blockIdx.y, k = blockIdx.x, tid = threadIdx.x;
    __shared__ _Float16 Ash[16 * TD];
    __shared__ _Float16 Wsh[256 * 32];
    __shared__ float gsh[TB];

    int a = active[t * TK + k];
    const float* xt = x + (size_t)t * TB * TD;

    stage_A_f32(xt, Ash, tid);

    // Gate dots: wave w handles batch row w; 8 contiguous floats per lane,
    // wave32 shuffle reduction.
    {
        int bb = tid >> 5, ln = tid & 31;
        int d0 = ln * 8;
        float acc = 0.f;
#pragma unroll
        for (int j = 0; j < 8; ++j)
            acc += xt[bb * TD + d0 + j] * gW[(size_t)a * TD + d0 + j];
#pragma unroll
        for (int o = 16; o > 0; o >>= 1) acc += __shfl_down(acc, o, 32);
        if (ln == 0) gsh[bb] = sigmoidf_(acc + gb[a]);
    }

    V8F c[2];
#pragma unroll
    for (int i = 0; i < 8; ++i) { c[0].f[i] = 0.f; c[1].f[i] = 0.f; }

    gemm_xWT(Ash, fcW + (size_t)a * TD * TD, Wsh, c, tid);

    int lane = tid & 31, wv = tid >> 5, hl = lane >> 4, n0 = lane & 15;
#pragma unroll
    for (int s = 0; s < 2; ++s) {
        int o = wv * 32 + s * 16 + n0;
        float bias = fcb[(size_t)a * TD + o];
#pragma unroll
        for (int r = 0; r < 8; ++r) {
            int b = r + hl * 8;
            if (b < TB) {
                float h  = c[s].f[r] + bias;
                float st = gelu_(h) * gsh[b] * wbuf[((size_t)t * TB + b) * TK + k];
                size_t off = (((size_t)t * TK + k) * TB + b) * TD + o;
                states[off]   = st;
                states_h[off] = (_Float16)st;
            }
        }
    }
}

// ---------------------------------------------------------------------------
// 4) Edge pass over compacted active edges: sig = states[src] @ conn^T,
//    act = gelu(sig @ fc[dst]^T + b) * sigmoid(sig.gate[dst]); atomic into nxt[dst].
// ---------------------------------------------------------------------------
__global__ __launch_bounds__(256) void edge_kernel(const _Float16* __restrict__ states_h,
                                                   const float* __restrict__ connW,
                                                   const float* __restrict__ fcW,
                                                   const float* __restrict__ fcb,
                                                   const float* __restrict__ gW,
                                                   const float* __restrict__ gb,
                                                   const int* __restrict__ aeIdx,
                                                   const int* __restrict__ aeCnt,
                                                   const int* __restrict__ posArr,
                                                   const int* __restrict__ esrc,
                                                   const int* __restrict__ edst,
                                                   float* __restrict__ nxt,
                                                   int E)
{
    int t = blockIdx.y;
    if ((int)blockIdx.x >= aeCnt[t]) return;   // uniform early exit, no syncs yet
    int e  = aeIdx[(size_t)t * E + blockIdx.x];
    int sn = esrc[e], dn = edst[e];
    int ks = posArr[t * TN + sn];
    int kd = posArr[t * TN + dn];
    int tid = threadIdx.x;

    __shared__ _Float16 Ash[16 * TD];
    __shared__ _Float16 Wsh[256 * 32];
    __shared__ float gsh[TB];

    // Stage A = states_h[t][ks] (f16 source, b64 copies; rows 8..15 zero).
    const _Float16* st = states_h + ((size_t)t * TK + ks) * TB * TD;
    for (int i = tid * 4; i < 16 * TD; i += 256 * 4) {
        int r = i >> 8;
        unsigned long long v = (r < TB) ? *(const unsigned long long*)(st + i) : 0ULL;
        *(unsigned long long*)(Ash + i) = v;
    }

    V8F c[2];
#pragma unroll
    for (int i = 0; i < 8; ++i) { c[0].f[i] = 0.f; c[1].f[i] = 0.f; }
    gemm_xWT(Ash, connW + (size_t)e * TD * TD, Wsh, c, tid);   // sig

    // Write sig back into Ash as f16 (rows 8..15 are exact zeros).
    int lane = tid & 31, wv = tid >> 5, hl = lane >> 4, n0 = lane & 15;
#pragma unroll
    for (int s = 0; s < 2; ++s) {
        int o = wv * 32 + s * 16 + n0;
#pragma unroll
        for (int r = 0; r < 8; ++r) {
            int b = r + hl * 8;
            Ash[b * TD + o] = (_Float16)c[s].f[r];
        }
    }
    __syncthreads();

    // Gate dots on sig: wave w -> batch row w, shuffle reduce.
    {
        int bb = tid >> 5, ln = tid & 31;
        int d0 = ln * 8;
        float acc = 0.f;
#pragma unroll
        for (int j = 0; j < 8; ++j)
            acc += (float)Ash[bb * TD + d0 + j] * gW[(size_t)dn * TD + d0 + j];
#pragma unroll
        for (int o = 16; o > 0; o >>= 1) acc += __shfl_down(acc, o, 32);
        if (ln == 0) gsh[bb] = sigmoidf_(acc + gb[dn]);
    }
    __syncthreads();

    V8F c2[2];
#pragma unroll
    for (int i = 0; i < 8; ++i) { c2[0].f[i] = 0.f; c2[1].f[i] = 0.f; }
    gemm_xWT(Ash, fcW + (size_t)dn * TD * TD, Wsh, c2, tid);   // hh

#pragma unroll
    for (int s = 0; s < 2; ++s) {
        int o = wv * 32 + s * 16 + n0;
        float bias = fcb[(size_t)dn * TD + o];
#pragma unroll
        for (int r = 0; r < 8; ++r) {
            int b = r + hl * 8;
            if (b < TB) {
                float h = c2[s].f[r] + bias;
                float a = gelu_(h) * gsh[b];
                atomicAdd(&nxt[(((size_t)t * TK + kd) * TB + b) * TD + o], a);
            }
        }
    }
}

// ---------------------------------------------------------------------------
// 5) Combine + mean over k + LayerNorm -> y_h (f16, [B*S][D] row = b*S+t)
// ---------------------------------------------------------------------------
__global__ __launch_bounds__(256) void combine_ln_kernel(const float* __restrict__ states,
                                                         const float* __restrict__ nxt,
                                                         const float* __restrict__ lng,
                                                         const float* __restrict__ lnb,
                                                         _Float16* __restrict__ y_h)
{
    int t = blockIdx.x, b = blockIdx.y, d = threadIdx.x;
    float acc = 0.f;
    for (int k = 0; k < TK; ++k) {
        size_t o = (((size_t)t * TK + k) * TB + b) * TD + d;
        acc += states[o] + 0.5f * nxt[o];
    }
    acc *= (1.0f / (float)TK);

    __shared__ float red[256];
    __shared__ float mu_s, var_s;
    red[d] = acc; __syncthreads();
    for (int off = 128; off > 0; off >>= 1) { if (d < off) red[d] += red[d + off]; __syncthreads(); }
    if (d == 0) mu_s = red[0] * (1.0f / TD);
    __syncthreads();
    float cd = acc - mu_s;
    red[d] = cd * cd; __syncthreads();
    for (int off = 128; off > 0; off >>= 1) { if (d < off) red[d] += red[d + off]; __syncthreads(); }
    if (d == 0) var_s = red[0] * (1.0f / TD);
    __syncthreads();
    float yv = cd * rsqrtf(var_s + 1e-5f) * lng[d] + lnb[d];
    y_h[((size_t)b * TS + t) * TD + d] = (_Float16)yv;
}

// ---------------------------------------------------------------------------
// 6) Vocab projection: logits[row][v] = y[row] . out_W[v] + out_b[v]
//    Block covers 16 rows x 128 cols; 8 waves, one 16x16 WMMA tile pair each.
// ---------------------------------------------------------------------------
__global__ __launch_bounds__(256) void proj_kernel(const _Float16* __restrict__ y_h,
                                                   const float* __restrict__ outW,
                                                   const float* __restrict__ outb,
                                                   float* __restrict__ out)
{
    int nc = blockIdx.x;   // column chunk of 128
    int mt = blockIdx.y;   // row tile of 16
    int tid = threadIdx.x;
    __shared__ _Float16 Ash[16 * TD];
    __shared__ _Float16 Wsh[128 * 32];

    // Stage A (f16 -> f16, b64 copies).
    {
        const _Float16* src = y_h + (size_t)mt * 16 * TD;
        for (int i = tid * 4; i < 16 * TD; i += 256 * 4)
            *(unsigned long long*)(Ash + i) = *(const unsigned long long*)(src + i);
    }

    int lane = tid & 31, wv = tid >> 5, m = lane & 15, hl = lane >> 4;
    int obase = nc * 128;

    V8F c[2];
#pragma unroll
    for (int i = 0; i < 8; ++i) { c[0].f[i] = 0.f; c[1].f[i] = 0.f; }

    for (int kb = 0; kb < TD; kb += 32) {
        __syncthreads();
        for (int i = tid * 4; i < 128 * 32; i += 256 * 4) {
            int o  = obase + (i >> 5);
            int kk = i & 31;
            H4 p;
            if (o < TV) {
                const float* src = outW + (size_t)o * TD + kb + kk;
                if (kk == 0 && kb + 32 < TD) __builtin_prefetch(src + 32, 0, 3);
                float4 w4 = *(const float4*)src;
                p.h[0] = (_Float16)w4.x; p.h[1] = (_Float16)w4.y;
                p.h[2] = (_Float16)w4.z; p.h[3] = (_Float16)w4.w;
            } else {
                p.u = 0ULL;
            }
            *(unsigned long long*)(Wsh + i) = p.u;
        }
        __syncthreads();

        V16H a;
#pragma unroll
        for (int v = 0; v < 8; ++v) {
            int k0 = ((v < 4) ? (2 * v) : (16 + 2 * (v - 4))) + (hl ? 8 : 0);
            a.h[2 * v]     = Ash[m * TD + kb + k0];
            a.h[2 * v + 1] = Ash[m * TD + kb + k0 + 1];
        }
#pragma unroll
        for (int s = 0; s < 2; ++s) {
            int nt = wv * 2 + s;
            V16H b;
#pragma unroll
            for (int v = 0; v < 8; ++v) {
                int k = ((v < 4) ? (2 * v) : (16 + 2 * (v - 4))) + (hl ? 8 : 0);
                int o = nt * 16 + (lane & 15);
                b.h[2 * v]     = Wsh[o * 32 + k];
                b.h[2 * v + 1] = Wsh[o * 32 + k + 1];
            }
            c[s].v = __builtin_amdgcn_wmma_f32_16x16x32_f16(
                false, a.v, false, b.v, (short)0, c[s].v, false, false);
        }
    }

#pragma unroll
    for (int s = 0; s < 2; ++s) {
        int o = obase + wv * 32 + s * 16 + (lane & 15);
        if (o < TV) {
            float bias = outb[o];
#pragma unroll
            for (int r = 0; r < 8; ++r) {
                int row = mt * 16 + r + hl * 8;
                out[(size_t)row * TV + o] = c[s].f[r] + bias;
            }
        }
    }
}

// ---------------------------------------------------------------------------
extern "C" void kernel_launch(void* const* d_in, const int* in_sizes, int n_in,
                              void* d_out, int out_size, void* d_ws, size_t ws_size,
                              hipStream_t stream)
{
    const int*   ids      = (const int*)  d_in[0];
    const float* emb      = (const float*)d_in[1];
    const float* posE     = (const float*)d_in[2];
    const float* routerW  = (const float*)d_in[3];
    const float* routerB  = (const float*)d_in[4];
    const float* fcW      = (const float*)d_in[5];
    const float* fcb      = (const float*)d_in[6];
    const float* gW       = (const float*)d_in[7];
    const float* gb       = (const float*)d_in[8];
    const float* connW    = (const float*)d_in[9];
    const float* lng      = (const float*)d_in[10];
    const float* lnb      = (const float*)d_in[11];
    const float* outW     = (const float*)d_in[12];
    const float* outb     = (const float*)d_in[13];
    const int*   esrc     = (const int*)  d_in[14];
    const int*   edst     = (const int*)  d_in[15];
    const int    E        = in_sizes[14];

    char* ws = (char*)d_ws;
    size_t off = 0;
    auto alloc = [&](size_t bytes) -> char* {
        char* p = ws + off;
        off += (bytes + 255) & ~(size_t)255;
        return p;
    };

    float*     x        = (float*)    alloc((size_t)TS * TB * TD * 4);
    float*     states   = (float*)    alloc((size_t)TS * TK * TB * TD * 4);
    _Float16*  states_h = (_Float16*) alloc((size_t)TS * TK * TB * TD * 2);
    float*     nxt      = (float*)    alloc((size_t)TS * TK * TB * TD * 4);
    float*     wbuf     = (float*)    alloc((size_t)TS * TB * TK * 4);
    int*       active   = (int*)      alloc((size_t)TS * TK * 4);
    int*       posArr   = (int*)      alloc((size_t)TS * TN * 4);
    int*       aeIdx    = (int*)      alloc((size_t)TS * E * 4);
    int*       aeCnt    = (int*)      alloc((size_t)TS * 4);
    _Float16*  y_h      = (_Float16*) alloc((size_t)TB * TS * TD * 2);
    (void)ws_size; (void)n_in; (void)out_size;

    hipMemsetAsync(nxt, 0, (size_t)TS * TK * TB * TD * 4, stream);

    embed_kernel<<<TB * TS, 256, 0, stream>>>(ids, emb, posE, x);

    router_kernel<<<TS, 256, 0, stream>>>(x, routerW, routerB, esrc, edst, E,
                                          active, wbuf, posArr, aeIdx, aeCnt);

    hop0_kernel<<<dim3(TK, TS), 256, 0, stream>>>(x, fcW, fcb, gW, gb,
                                                  active, wbuf, states, states_h);

    edge_kernel<<<dim3(E, TS), 256, 0, stream>>>(states_h, connW, fcW, fcb, gW, gb,
                                                 aeIdx, aeCnt, posArr, esrc, edst, nxt, E);

    combine_ln_kernel<<<dim3(TS, TB), 256, 0, stream>>>(states, nxt, lng, lnb, y_h);

    proj_kernel<<<dim3((TV + 127) / 128, (TB * TS) / 16), 256, 0, stream>>>(y_h, outW, outb,
                                                                            (float*)d_out);
}
